// StrokeARFMv2_12317966205192
// MI455X (gfx1250) — matrix operand order repaired and checked by reading, not compiled, not run on hardware
//
#include <hip/hip_runtime.h>
#include <hip/hip_bf16.h>
#include <math.h>

// ---------------------------------------------------------------------------
// Problem constants (from the reference)
// ---------------------------------------------------------------------------
#define B_   8192
#define T_   30
#define D_   256
#define H_   8
#define HD_  32
#define L_   6
#define SD_  8
#define FF_  1024
#define CF_  256
#define TEMB_ 32
#define ROWS_ (B_ * T_)          // 245760
#define KPAD_ 320                // head input 296 padded to 320 (mult of 32)
#define KCHUNK_ 256              // K staged into LDS per TDM transfer

typedef __attribute__((ext_vector_type(16))) __bf16 v16bf;
typedef __attribute__((ext_vector_type(8)))  __bf16 v8bf;
typedef __attribute__((ext_vector_type(8)))  float  v8f;
typedef __attribute__((ext_vector_type(4)))  unsigned int v4u;
typedef __attribute__((ext_vector_type(8)))  int   v8i;
typedef __attribute__((ext_vector_type(4)))  int   v4i;

// ---------------------------------------------------------------------------
// f32 -> bf16 conversion (weights / activations)
// ---------------------------------------------------------------------------
__global__ void cvt_f32_bf16(const float* __restrict__ s, __bf16* __restrict__ d, size_t n) {
    size_t i = (size_t)blockIdx.x * blockDim.x + threadIdx.x;
    size_t stride = (size_t)gridDim.x * blockDim.x;
    for (; i < n; i += stride) d[i] = (__bf16)s[i];
}

// f32 (N x Ks) -> bf16 (N x Kd) with zero right-pad (for fm_w1: 296 -> 320)
__global__ void cvt_pad_f32_bf16(const float* __restrict__ s, __bf16* __restrict__ d,
                                 int N, int Ks, int Kd) {
    size_t i = (size_t)blockIdx.x * blockDim.x + threadIdx.x;
    size_t n = (size_t)N * Kd;
    if (i >= n) return;
    int row = (int)(i / Kd), k = (int)(i % Kd);
    d[i] = (k < Ks) ? (__bf16)s[(size_t)row * Ks + k] : (__bf16)0.0f;
}

__global__ void zero_out(float* o) { o[0] = 0.0f; }

// ---------------------------------------------------------------------------
// Tensor Data Mover: DMA a KC x 64-row bf16 weight panel (row stride K elems)
// into LDS at byte offset ldsOff. Issued by one wave; tracked by TENSORcnt.
// D# layout per CDNA5 ISA ch.8 (group0: count/lds/global addr; group1: dims).
// ---------------------------------------------------------------------------
__device__ __forceinline__ void tdm_load_panel(const __bf16* gsrc, unsigned int ldsOff,
                                               int K, int N, int KC) {
    unsigned long long ga = (unsigned long long)(uintptr_t)gsrc;
    v4u g0;
    g0[0] = 1u;                                              // count=1 (valid user D#)
    g0[1] = ldsOff;                                          // lds_addr
    g0[2] = (unsigned int)(ga & 0xFFFFFFFFull);              // global_addr[31:0]
    g0[3] = (unsigned int)((ga >> 32) & 0x1FFFFFFull)        // global_addr[56:32]
          | (2u << 30);                                      // type = 2 ("image")
    v8i g1;
    g1[0] = (int)(1u << 16);                                 // data_size=1 -> 2 bytes
    g1[1] = (int)(((unsigned int)K & 0xFFFFu) << 16);        // tensor_dim0[15:0]
    g1[2] = (int)((((unsigned int)K >> 16) & 0xFFFFu)        // tensor_dim0[31:16]
          | (((unsigned int)N & 0xFFFFu) << 16));            // tensor_dim1[15:0]
    g1[3] = (int)((((unsigned int)N >> 16) & 0xFFFFu)        // tensor_dim1[31:16]
          | (((unsigned int)KC & 0xFFFFu) << 16));           // tile_dim0 = KC
    g1[4] = (int)64u;                                        // tile_dim1=64, tile_dim2=0
    g1[5] = (int)(unsigned int)K;                            // tensor_dim0_stride[31:0]
    g1[6] = 0;                                               // stride hi / dim1_stride lo
    g1[7] = 0;                                               // dim1_stride hi
    v4i gz; gz[0] = gz[1] = gz[2] = gz[3] = 0;
#if defined(__clang_major__) && (__clang_major__ >= 23)
    v8i gz8; gz8[0]=gz8[1]=gz8[2]=gz8[3]=gz8[4]=gz8[5]=gz8[6]=gz8[7]=0;
    __builtin_amdgcn_tensor_load_to_lds(g0, g1, gz, gz, gz8, 0);
#else
    __builtin_amdgcn_tensor_load_to_lds(g0, g1, gz, gz, 0);
#endif
}

// ---------------------------------------------------------------------------
// Generic bf16 WMMA GEMM:  out = act( A[MxK] * W[NxK]^T + bias ) (+ resid)
// Block = 256 threads (8 waves). Block tile 128(M) x 64(N); wave tile 16 x 64.
// B panel (64 rows x KCHUNK) staged into LDS by the Tensor Data Mover;
// A fragments stream from global (wave-private rows, 16 contiguous K per half-wave).
// Requires: M % 128 == 0, N % 64 == 0, K % 32 == 0 (holds for every GEMM here).
// act: 0 = none, 1 = silu
// ---------------------------------------------------------------------------
__global__ __launch_bounds__(256)
void gemm_bf16_wmma(const __bf16* __restrict__ A, const __bf16* __restrict__ W,
                    const float* __restrict__ bias, const float* __restrict__ resid,
                    float* __restrict__ out32, __bf16* __restrict__ out16,
                    int M, int N, int K, int act) {
    __shared__ __bf16 sB[64 * KCHUNK_];      // 32 KB weight panel

    const int wave = threadIdx.x >> 5;
    const int lane = threadIdx.x & 31;
    const int g    = lane >> 4;              // half-wave group (K-split for frags)
    const int ln   = lane & 15;
    const int m0   = blockIdx.y * 128 + wave * 16;
    const int n0   = blockIdx.x * 64;

    v8f acc0 = {}, acc1 = {}, acc2 = {}, acc3 = {};

    const __bf16* arow = A + (size_t)(m0 + ln) * K;

    for (int kc = 0; kc < K; kc += KCHUNK_) {
        const int KC = (K - kc) < KCHUNK_ ? (K - kc) : KCHUNK_;

        // ---- TDM: one wave DMAs the 64 x KC weight panel into LDS ----
        if (threadIdx.x < 32) {
            tdm_load_panel(W + (size_t)n0 * K + kc,
                           (unsigned int)(uintptr_t)&sB[0], K, N, KC);
            __builtin_amdgcn_s_wait_tensorcnt(0);
        }
        __syncthreads();

        for (int kb = 0; kb < KC; kb += 32) {
            // ---- A fragment: 16x32 bf16, row = m0+ln, 16 contiguous K per lane ----
            const __bf16* ap = arow + kc + kb + g * 16;
            v8bf alo = *(const v8bf*)(ap);
            v8bf ahi = *(const v8bf*)(ap + 8);
            v16bf af;
#pragma unroll
            for (int i = 0; i < 8; ++i) { af[i] = alo[i]; af[8 + i] = ahi[i]; }
            __builtin_prefetch(ap + 32, 0, 1);   // next K slab -> global_prefetch_b8

            // ---- B fragments from LDS: col = ln, 16 contiguous K per lane ----
#pragma unroll
            for (int j = 0; j < 4; ++j) {
                const __bf16* bp = &sB[(size_t)(j * 16 + ln) * KC + kb + g * 16];
                v16bf bfrag = *(const v16bf*)bp;
                switch (j) {
                    case 0: acc0 = __builtin_amdgcn_wmma_f32_16x16x32_bf16(false, af, false, bfrag, (short)0, acc0, false, false); break;
                    case 1: acc1 = __builtin_amdgcn_wmma_f32_16x16x32_bf16(false, af, false, bfrag, (short)0, acc1, false, false); break;
                    case 2: acc2 = __builtin_amdgcn_wmma_f32_16x16x32_bf16(false, af, false, bfrag, (short)0, acc2, false, false); break;
                    case 3: acc3 = __builtin_amdgcn_wmma_f32_16x16x32_bf16(false, af, false, bfrag, (short)0, acc3, false, false); break;
                }
            }
        }
        __syncthreads();      // all waves done reading panel before next DMA
    }

    // ---- epilogue: C/D layout -> m = m0 + g*8 + r, n = n0 + j*16 + ln ----
    const int mbase = m0 + g * 8;
#pragma unroll
    for (int j = 0; j < 4; ++j) {
        const v8f acc = (j == 0) ? acc0 : (j == 1) ? acc1 : (j == 2) ? acc2 : acc3;
        const int n = n0 + j * 16 + ln;
        const float bv = bias ? bias[n] : 0.0f;
#pragma unroll
        for (int r = 0; r < 8; ++r) {
            const int m = mbase + r;
            float v = acc[r] + bv;
            if (act == 1) v = v * (1.0f / (1.0f + __expf(-v)));   // silu
            const size_t o = (size_t)m * N + n;
            if (resid) v += resid[o];
            if (out32) out32[o] = v;
            if (out16) out16[o] = (__bf16)v;
        }
    }
}

// ---------------------------------------------------------------------------
// Stroke projection: x[b,t,:] = strokes[b,t,:] @ sp_w^T + sp_b + pos_emb[t]
// one block (256 thr) per (b,t) row
// ---------------------------------------------------------------------------
__global__ __launch_bounds__(256)
void stroke_proj(const float* __restrict__ strokes, const float* __restrict__ sp_w,
                 const float* __restrict__ sp_b, const float* __restrict__ pos_emb,
                 float* __restrict__ x) {
    const int row = blockIdx.x;
    const int d   = threadIdx.x;
    const int t   = row % T_;
    const float* s = strokes + (size_t)row * SD_;
    float acc = sp_b[d];
#pragma unroll
    for (int i = 0; i < SD_; ++i) acc += s[i] * sp_w[d * SD_ + i];
    x[(size_t)row * D_ + d] = acc + pos_emb[t * D_ + d];
}

// ---------------------------------------------------------------------------
// LN + adaLN modulate: out16[row] = (1+sc[b]) * ln(x[row]) + sh[b]   (bf16)
// one block (256 thr) per (b,t) row; LDS tree reduction for mean/var
// ---------------------------------------------------------------------------
__global__ __launch_bounds__(256)
void ln_modulate(const float* __restrict__ x, const float* __restrict__ ss,
                 __bf16* __restrict__ out) {
    const int row = blockIdx.x;
    const int tid = threadIdx.x;
    const int b   = row / T_;
    __shared__ float s1[256];
    __shared__ float s2[256];
    const float v = x[(size_t)row * D_ + tid];
    s1[tid] = v; s2[tid] = v * v;
    __syncthreads();
    for (int off = 128; off; off >>= 1) {
        if (tid < off) { s1[tid] += s1[tid + off]; s2[tid] += s2[tid + off]; }
        __syncthreads();
    }
    const float mean = s1[0] * (1.0f / D_);
    const float var  = s2[0] * (1.0f / D_) - mean * mean;
    const float nrm  = (v - mean) * rsqrtf(var + 1e-5f);
    const float sc   = ss[(size_t)b * (2 * D_) + tid];
    const float sh   = ss[(size_t)b * (2 * D_) + D_ + tid];
    out[(size_t)row * D_ + tid] = (__bf16)((1.0f + sc) * nrm + sh);
}

// ---------------------------------------------------------------------------
// Causal attention, one wave32 per (b, h). qkv is bf16 (B*T, 768);
// K/V for the head staged in LDS, lane = query position t.
// ---------------------------------------------------------------------------
__global__ __launch_bounds__(32)
void attention(const __bf16* __restrict__ qkv, __bf16* __restrict__ out) {
    const int bh   = blockIdx.x;
    const int b    = bh >> 3;
    const int h    = bh & 7;
    const int lane = threadIdx.x;

    __shared__ float ks[T_ * HD_];
    __shared__ float vs[T_ * HD_];
    for (int j = 0; j < T_; ++j) {
        const size_t base = ((size_t)b * T_ + j) * (3 * D_) + h * HD_;
        ks[j * HD_ + lane] = (float)qkv[base + D_ + lane];
        vs[j * HD_ + lane] = (float)qkv[base + 2 * D_ + lane];
    }
    __syncthreads();

    const int t = lane;
    if (t < T_) {
        float q[HD_];
        const size_t qb = ((size_t)b * T_ + t) * (3 * D_) + h * HD_;
#pragma unroll
        for (int c = 0; c < HD_; ++c) q[c] = (float)qkv[qb + c];

        const float scale = 0.17677669529663687f;   // 1/sqrt(32)
        float lg[T_];
        float mx = -1e30f;
        for (int j = 0; j <= t; ++j) {
            float a = 0.0f;
#pragma unroll
            for (int c = 0; c < HD_; ++c) a += q[c] * ks[j * HD_ + c];
            a *= scale;
            lg[j] = a;
            mx = a > mx ? a : mx;
        }
        float denom = 0.0f;
        for (int j = 0; j <= t; ++j) { lg[j] = __expf(lg[j] - mx); denom += lg[j]; }
        const float inv = 1.0f / denom;
        const size_t ob = ((size_t)b * T_ + t) * D_ + h * HD_;
        for (int c = 0; c < HD_; ++c) {
            float a = 0.0f;
            for (int j = 0; j <= t; ++j) a += lg[j] * vs[j * HD_ + c];
            out[ob + c] = (__bf16)(a * inv);
        }
    }
}

// ---------------------------------------------------------------------------
// Final LN on last token + build flow-matching input xin (bf16, 320 wide):
//   [0:256)=ln(x[b,T-1])*g+b, [256:264)=at, [264:296)=temb, [296:320)=0
// also stores u = a_tar - a_src (f32). One block (256 thr) per batch b.
// ---------------------------------------------------------------------------
__global__ __launch_bounds__(256)
void build_xin(const float* __restrict__ x, const float* __restrict__ on_g,
               const float* __restrict__ on_b, const float* __restrict__ a_tar,
               const float* __restrict__ a_src, const float* __restrict__ tvec,
               __bf16* __restrict__ xin, float* __restrict__ u) {
    const int b   = blockIdx.x;
    const int tid = threadIdx.x;
    __shared__ float s1[256];
    __shared__ float s2[256];
    const float v = x[((size_t)b * T_ + (T_ - 1)) * D_ + tid];
    s1[tid] = v; s2[tid] = v * v;
    __syncthreads();
    for (int off = 128; off; off >>= 1) {
        if (tid < off) { s1[tid] += s1[tid + off]; s2[tid] += s2[tid + off]; }
        __syncthreads();
    }
    const float mean = s1[0] * (1.0f / D_);
    const float var  = s2[0] * (1.0f / D_) - mean * mean;
    const float hval = (v - mean) * rsqrtf(var + 1e-5f) * on_g[tid] + on_b[tid];

    __bf16* xr = xin + (size_t)b * KPAD_;
    xr[tid] = (__bf16)hval;

    const float tb = tvec[b];
    if (tid < SD_) {
        const float as = a_src[b * SD_ + tid];
        const float atr = a_tar[b * SD_ + tid];
        xr[D_ + tid] = (__bf16)((1.0f - tb) * as + tb * atr);
        u[b * SD_ + tid] = atr - as;
    }
    if (tid < TEMB_) {
        const int half = TEMB_ / 2;
        const int i = (tid < half) ? tid : (tid - half);
        const float f = __expf(-logf(10000.0f) * (float)i / (float)(half - 1));
        const float arg = tb * f;
        xr[D_ + SD_ + tid] = (__bf16)((tid < half) ? __sinf(arg) : __cosf(arg));
    }
    if (tid < (KPAD_ - D_ - SD_ - TEMB_)) xr[D_ + SD_ + TEMB_ + tid] = (__bf16)0.0f;
}

// ---------------------------------------------------------------------------
// Final tiny GEMM (512 -> 8) fused with MSE loss reduction.
// thread = (b, n) pair; block reduce; atomicAdd scalar.
// ---------------------------------------------------------------------------
__global__ __launch_bounds__(256)
void head_loss(const __bf16* __restrict__ hid, const __bf16* __restrict__ w3,
               const float* __restrict__ b3, const float* __restrict__ u,
               float* __restrict__ out) {
    const int idx = blockIdx.x * 256 + threadIdx.x;
    const int b = idx >> 3;
    const int n = idx & 7;
    const __bf16* hr = hid + (size_t)b * 512;
    const __bf16* wr = w3 + (size_t)n * 512;
    float a = b3[n];
#pragma unroll 8
    for (int k = 0; k < 512; ++k) a += (float)hr[k] * (float)wr[k];
    const float dv = a - u[idx];
    __shared__ float red[256];
    red[threadIdx.x] = dv * dv;
    __syncthreads();
    for (int off = 128; off; off >>= 1) {
        if (threadIdx.x < off) red[threadIdx.x] += red[threadIdx.x + off];
        __syncthreads();
    }
    if (threadIdx.x == 0) atomicAdd(out, red[0] * (1.0f / (float)(B_ * SD_)));
}

// ---------------------------------------------------------------------------
// Host orchestration
// ---------------------------------------------------------------------------
static inline size_t align256(size_t x) { return (x + 255) & ~(size_t)255; }

extern "C" void kernel_launch(void* const* d_in, const int* in_sizes, int n_in,
                              void* d_out, int out_size, void* d_ws, size_t ws_size,
                              hipStream_t stream) {
    // ---- inputs (setup_inputs order) ----
    const float* strokes  = (const float*)d_in[0];
    const float* canvas   = (const float*)d_in[1];
    const float* a_tar    = (const float*)d_in[2];
    const float* a_src    = (const float*)d_in[3];
    const float* tvec     = (const float*)d_in[4];
    const float* ce_w1    = (const float*)d_in[5];
    const float* ce_b1    = (const float*)d_in[6];
    const float* ce_w2    = (const float*)d_in[7];
    const float* ce_b2    = (const float*)d_in[8];
    const float* sp_w     = (const float*)d_in[9];
    const float* sp_b     = (const float*)d_in[10];
    const float* pos_emb  = (const float*)d_in[11];
    const float* adaln1_w = (const float*)d_in[12];
    const float* adaln1_b = (const float*)d_in[13];
    const float* inproj_w = (const float*)d_in[14];
    const float* inproj_b = (const float*)d_in[15];
    const float* outproj_w= (const float*)d_in[16];
    const float* outproj_b= (const float*)d_in[17];
    const float* adaln2_w = (const float*)d_in[18];
    const float* adaln2_b = (const float*)d_in[19];
    const float* ff1_w    = (const float*)d_in[20];
    const float* ff1_b    = (const float*)d_in[21];
    const float* ff2_w    = (const float*)d_in[22];
    const float* ff2_b    = (const float*)d_in[23];
    const float* on_g     = (const float*)d_in[24];
    const float* on_b     = (const float*)d_in[25];
    const float* fm_w1    = (const float*)d_in[26];
    const float* fm_b1    = (const float*)d_in[27];
    const float* fm_w2    = (const float*)d_in[28];
    const float* fm_b2    = (const float*)d_in[29];
    const float* fm_w3    = (const float*)d_in[30];
    const float* fm_b3    = (const float*)d_in[31];
    float* out = (float*)d_out;

    // ---- workspace carve-up ----
    char* p = (char*)d_ws;
    auto alloc = [&](size_t bytes) -> char* { char* r = p; p += align256(bytes); return r; };

    float*  xf    = (float*)alloc((size_t)ROWS_ * D_ * 4);        // residual stream f32
    __bf16* x2h   = (__bf16*)alloc((size_t)ROWS_ * D_ * 2);       // LN-modulated (GEMM A)
    __bf16* midh  = (__bf16*)alloc((size_t)ROWS_ * FF_ * 2);      // qkv(768) / ff hidden(1024)
    __bf16* aoh   = (__bf16*)alloc((size_t)ROWS_ * D_ * 2);       // attention out
    float*  ss    = (float*)alloc((size_t)B_ * 2 * D_ * 4);       // adaLN scale/shift
    __bf16* cfh   = (__bf16*)alloc((size_t)B_ * 384 * 2);         // canvas feat bf16
    __bf16* cmid  = (__bf16*)alloc((size_t)B_ * 512 * 2);         // canvas hidden
    __bf16* ch    = (__bf16*)alloc((size_t)B_ * CF_ * 2);         // conditioning c bf16
    __bf16* xin   = (__bf16*)alloc((size_t)B_ * KPAD_ * 2);       // head input (padded)
    __bf16* hid1  = (__bf16*)alloc((size_t)B_ * 512 * 2);
    __bf16* hid2  = (__bf16*)alloc((size_t)B_ * 512 * 2);
    float*  uvec  = (float*)alloc((size_t)B_ * SD_ * 4);

    __bf16* ce_w1h   = (__bf16*)alloc((size_t)512 * 384 * 2);
    __bf16* ce_w2h   = (__bf16*)alloc((size_t)256 * 512 * 2);
    __bf16* adaln1h  = (__bf16*)alloc((size_t)L_ * 512 * 256 * 2);
    __bf16* inprojh  = (__bf16*)alloc((size_t)L_ * 768 * 256 * 2);
    __bf16* outprojh = (__bf16*)alloc((size_t)L_ * 256 * 256 * 2);
    __bf16* adaln2h  = (__bf16*)alloc((size_t)L_ * 512 * 256 * 2);
    __bf16* ff1h     = (__bf16*)alloc((size_t)L_ * 1024 * 256 * 2);
    __bf16* ff2h     = (__bf16*)alloc((size_t)L_ * 256 * 1024 * 2);
    __bf16* fmw1h    = (__bf16*)alloc((size_t)512 * KPAD_ * 2);
    __bf16* fmw2h    = (__bf16*)alloc((size_t)512 * 512 * 2);
    __bf16* fmw3h    = (__bf16*)alloc((size_t)8 * 512 * 2);
    (void)ws_size; (void)in_sizes; (void)n_in; (void)out_size;

    auto cvt = [&](const float* s, __bf16* d, size_t n) {
        int blocks = (int)((n + 255) / 256);
        if (blocks > 4096) blocks = 4096;
        hipLaunchKernelGGL(cvt_f32_bf16, dim3(blocks), dim3(256), 0, stream, s, d, n);
    };

    // ---- 0: output init + weight conversion ----
    hipLaunchKernelGGL(zero_out, dim3(1), dim3(1), 0, stream, out);
    cvt(canvas,   cfh,      (size_t)B_ * 384);
    cvt(ce_w1,    ce_w1h,   (size_t)512 * 384);
    cvt(ce_w2,    ce_w2h,   (size_t)256 * 512);
    cvt(adaln1_w, adaln1h,  (size_t)L_ * 512 * 256);
    cvt(inproj_w, inprojh,  (size_t)L_ * 768 * 256);
    cvt(outproj_w,outprojh, (size_t)L_ * 256 * 256);
    cvt(adaln2_w, adaln2h,  (size_t)L_ * 512 * 256);
    cvt(ff1_w,    ff1h,     (size_t)L_ * 1024 * 256);
    cvt(ff2_w,    ff2h,     (size_t)L_ * 256 * 1024);
    cvt(fm_w2,    fmw2h,    (size_t)512 * 512);
    cvt(fm_w3,    fmw3h,    (size_t)8 * 512);
    {
        size_t n = (size_t)512 * KPAD_;
        hipLaunchKernelGGL(cvt_pad_f32_bf16, dim3((unsigned)((n + 255) / 256)), dim3(256),
                           0, stream, fm_w1, fmw1h, 512, 296, KPAD_);
    }

    auto gemm = [&](const __bf16* A, const __bf16* W, const float* bias, const float* resid,
                    float* o32, __bf16* o16, int M, int N, int K, int act) {
        dim3 grid((unsigned)(N / 64), (unsigned)(M / 128));
        hipLaunchKernelGGL(gemm_bf16_wmma, grid, dim3(256), 0, stream,
                           A, W, bias, resid, o32, o16, M, N, K, act);
    };

    // ---- 1: canvas encoder c = silu(canvas @ ce_w1^T) @ ce_w2^T ----
    gemm(cfh,  ce_w1h, ce_b1, nullptr, nullptr, cmid, B_, 512, 384, 1);
    gemm(cmid, ce_w2h, ce_b2, nullptr, nullptr, ch,   B_, CF_, 512, 0);

    // ---- 2: stroke projection + positional embedding -> xf ----
    hipLaunchKernelGGL(stroke_proj, dim3(ROWS_), dim3(D_), 0, stream,
                       strokes, sp_w, sp_b, pos_emb, xf);

    // ---- 3: transformer layers ----
    for (int l = 0; l < L_; ++l) {
        // adaLN-1
        gemm(ch, adaln1h + (size_t)l * 512 * 256, adaln1_b + (size_t)l * 512,
             nullptr, ss, nullptr, B_, 512, 256, 0);
        hipLaunchKernelGGL(ln_modulate, dim3(ROWS_), dim3(D_), 0, stream, xf, ss, x2h);
        // qkv projection
        gemm(x2h, inprojh + (size_t)l * 768 * 256, inproj_b + (size_t)l * 768,
             nullptr, nullptr, midh, ROWS_, 3 * D_, 256, 0);
        // attention
        hipLaunchKernelGGL(attention, dim3(B_ * H_), dim3(32), 0, stream, midh, aoh);
        // output projection + residual
        gemm(aoh, outprojh + (size_t)l * 256 * 256, outproj_b + (size_t)l * 256,
             xf, xf, nullptr, ROWS_, D_, 256, 0);
        // adaLN-2
        gemm(ch, adaln2h + (size_t)l * 512 * 256, adaln2_b + (size_t)l * 512,
             nullptr, ss, nullptr, B_, 512, 256, 0);
        hipLaunchKernelGGL(ln_modulate, dim3(ROWS_), dim3(D_), 0, stream, xf, ss, x2h);
        // feed-forward + residual
        gemm(x2h, ff1h + (size_t)l * 1024 * 256, ff1_b + (size_t)l * 1024,
             nullptr, nullptr, midh, ROWS_, FF_, 256, 1);
        gemm(midh, ff2h + (size_t)l * 256 * 1024, ff2_b + (size_t)l * 256,
             xf, xf, nullptr, ROWS_, D_, 1024, 0);
    }

    // ---- 4: final LN + flow-matching input ----
    hipLaunchKernelGGL(build_xin, dim3(B_), dim3(D_), 0, stream,
                       xf, on_g, on_b, a_tar, a_src, tvec, xin, uvec);

    // ---- 5: head MLP ----
    gemm(xin,  fmw1h, fm_b1, nullptr, nullptr, hid1, B_, 512, KPAD_, 1);
    gemm(hid1, fmw2h, fm_b2, nullptr, nullptr, hid2, B_, 512, 512, 1);

    // ---- 6: final projection + MSE reduction ----
    hipLaunchKernelGGL(head_loss, dim3((B_ * SD_) / 256), dim3(256), 0, stream,
                       hid2, fmw3h, fm_b3, uvec, out);
}
